// QMLP_48893907697790
// MI455X (gfx1250) — compile-verified, matrix operand order checked by reading, not run on hardware
//
#include <hip/hip_runtime.h>
#include <hip/hip_bf16.h>
#include <cstdint>

typedef __attribute__((ext_vector_type(8))) int v8i;

#define B_SZ   4096
#define F_IN   1960
#define F_PAD  2048
#define HID    4096
#define C_OUT  12
#define C_PAD  16
#define BM     128
#define BN     128
#define BK     64
#define AP     72      // LDS row pitch (64B tile row + 8B pad -> conflict-free ds_load_b64)

// ---------------------------------------------------------------------------
// Fragment load from an LDS (or global) K-major int8 row, following the CDNA5
// 8-bit A/B 16x64 VGPR layout: lanes 0-15 hold K {0-7,16-23,32-39,48-55},
// lanes 16-31 hold K {8-15,24-31,40-47,56-63}  (lhi = 0 or 8).
// ---------------------------------------------------------------------------
__device__ __forceinline__ v8i frag_ld(const int8_t* base, int lhi) {
  const int8_t* p = base + lhi;
  unsigned long long q0 = *(const unsigned long long*)(p);
  unsigned long long q1 = *(const unsigned long long*)(p + 16);
  unsigned long long q2 = *(const unsigned long long*)(p + 32);
  unsigned long long q3 = *(const unsigned long long*)(p + 48);
  v8i r;
  r[0] = (int)(unsigned)q0;  r[1] = (int)(unsigned)(q0 >> 32);
  r[2] = (int)(unsigned)q1;  r[3] = (int)(unsigned)(q1 >> 32);
  r[4] = (int)(unsigned)q2;  r[5] = (int)(unsigned)(q2 >> 32);
  r[6] = (int)(unsigned)q3;  r[7] = (int)(unsigned)(q3 >> 32);
  return r;
}

// ---------------------------------------------------------------------------
// Per-output-channel symmetric 4-bit weight quantization (one block per row),
// zero-padding K to Kpad so WMMA K-chunks of 64 need no tail handling.
// ---------------------------------------------------------------------------
__global__ void qw_rows(const float* __restrict__ W, int8_t* __restrict__ Wq,
                        float* __restrict__ scales, int K, int Kpad, float qn) {
  const int row = blockIdx.x;
  const float* wr = W + (size_t)row * K;
  __shared__ float red[256];
  float mx = 0.f;
  for (int k = threadIdx.x; k < K; k += 256) mx = fmaxf(mx, fabsf(wr[k]));
  red[threadIdx.x] = mx;
  __syncthreads();
  for (int s = 128; s > 0; s >>= 1) {
    if (threadIdx.x < s) red[threadIdx.x] = fmaxf(red[threadIdx.x], red[threadIdx.x + s]);
    __syncthreads();
  }
  const float scale = fmaxf(red[0] / qn, 1e-8f);
  if (threadIdx.x == 0) scales[row] = scale;
  int8_t* out = Wq + (size_t)row * Kpad;
  for (int k = threadIdx.x; k < Kpad; k += 256) {
    float q = 0.f;
    if (k < K) q = fminf(fmaxf(rintf(wr[k] / scale), -qn), qn);
    out[k] = (int8_t)(int)q;
  }
}

// Per-tensor 8-bit quant for the output layer; rows padded 12 -> 16 with zeros.
__global__ void qw_tensor(const float* __restrict__ W, int8_t* __restrict__ Wq,
                          float* __restrict__ s_out, int rows, int K, int rowsPad,
                          float qn) {
  __shared__ float red[256];
  const int n = rows * K;
  float mx = 0.f;
  for (int i = threadIdx.x; i < n; i += 256) mx = fmaxf(mx, fabsf(W[i]));
  red[threadIdx.x] = mx;
  __syncthreads();
  for (int s = 128; s > 0; s >>= 1) {
    if (threadIdx.x < s) red[threadIdx.x] = fmaxf(red[threadIdx.x], red[threadIdx.x + s]);
    __syncthreads();
  }
  const float scale = fmaxf(red[0] / qn, 1e-8f);
  if (threadIdx.x == 0) s_out[0] = scale;
  for (int i = threadIdx.x; i < rowsPad * K; i += 256) {
    float q = 0.f;
    if (i < n) q = fminf(fmaxf(rintf(W[i] / scale), -qn), qn);
    Wq[i] = (int8_t)(int)q;
  }
}

// Shifted-uint8 input quant -> signed int8 (r - 128), zero-padded 1960 -> 2048.
__global__ void qin(const float* __restrict__ x, int8_t* __restrict__ aq,
                    const float* __restrict__ s_ptr) {
  const int idx = blockIdx.x * 256 + threadIdx.x;
  const int row = idx / F_PAD, k = idx - row * F_PAD;
  const float s = s_ptr[0];
  int8_t v = 0;
  if (k < F_IN) {
    float r = fminf(fmaxf(rintf(x[(size_t)row * F_IN + k] / s + 128.f), 0.f), 255.f);
    v = (int8_t)((int)r - 128);
  }
  aq[(size_t)row * F_PAD + k] = v;
}

// Fold BN + scales: cA[n] = sa*sw[n]*g[n]*rsqrt(v[n]+eps); cB[n] = b[n]-m[n]*alpha.
__global__ void coef(const float* __restrict__ sw, const float* __restrict__ g,
                     const float* __restrict__ b, const float* __restrict__ m,
                     const float* __restrict__ v, const float* __restrict__ sa,
                     float* __restrict__ cA, float* __restrict__ cB) {
  const int n = blockIdx.x * 256 + threadIdx.x;
  const float alpha = g[n] * rsqrtf(v[n] + 1e-5f);
  cA[n] = sa[0] * sw[n] * alpha;
  cB[n] = b[n] - m[n] * alpha;
}

// ---------------------------------------------------------------------------
// Fused int8 GEMM (C = A[M,K] x B[N,K]^T, int32 accum) + BN + ReLU + 4-bit
// requant. 256 threads = 8 waves; block tile 128x128x64; wave tile 64x32.
// Tiles staged with global_load_async_to_lds_b128, double-buffered, tracked
// with ASYNCcnt; compute with v_wmma_i32_16x16x64_iu8.
// ---------------------------------------------------------------------------
template <bool ASIGN>
__global__ __launch_bounds__(256) void gemm_iu8_bn_relu_q4(
    const int8_t* __restrict__ A, const int8_t* __restrict__ Bq,
    const float* __restrict__ cA, const float* __restrict__ cB,
    const float* __restrict__ s_next, int8_t* __restrict__ Out,
    int K, int ldo) {
  __shared__ int8_t As[2][BM * AP];
  __shared__ int8_t Bs[2][BN * AP];

  const int tid  = threadIdx.x;
  const int lane = tid & 31;
  const int wave = tid >> 5;
  const int wm   = wave >> 2;           // 0..1 -> 64 M rows
  const int wn   = wave & 3;            // 0..3 -> 32 N cols
  const int lrow = lane & 15;
  const int lhi  = (lane >> 4) << 3;    // 0 or 8
  const int bm   = blockIdx.x * BM;
  const int bn   = blockIdx.y * BN;

  const unsigned ldsA = (unsigned)(uintptr_t)(&As[0][0]);
  const unsigned ldsB = (unsigned)(uintptr_t)(&Bs[0][0]);
  const unsigned bufStride = BM * AP;

  v8i zero = {0, 0, 0, 0, 0, 0, 0, 0};
  v8i acc[4][2];
#pragma unroll
  for (int i = 0; i < 4; ++i)
#pragma unroll
    for (int j = 0; j < 2; ++j) acc[i][j] = zero;

  // 16KB/tile-pair per K-step: each thread issues 2 A + 2 B async b128 loads.
  auto issue = [&](int buf, int kk) {
    const int k0 = kk * BK;
#pragma unroll
    for (int i = 0; i < 2; ++i) {
      const int c   = tid * 2 + i;      // 0..511 16-byte chunks
      const int row = c >> 2;           // 4 chunks per 64B row
      const int col = (c & 3) << 4;
      unsigned ga = (unsigned)((bm + row) * K + k0 + col);
      unsigned la = ldsA + (unsigned)buf * bufStride + (unsigned)(row * AP + col);
      asm volatile("global_load_async_to_lds_b128 %0, %1, %2 offset:0"
                   :: "v"(la), "v"(ga), "s"(A) : "memory");
      unsigned gb = (unsigned)((bn + row) * K + k0 + col);
      unsigned lb = ldsB + (unsigned)buf * bufStride + (unsigned)(row * AP + col);
      asm volatile("global_load_async_to_lds_b128 %0, %1, %2 offset:0"
                   :: "v"(lb), "v"(gb), "s"(Bq) : "memory");
    }
  };

  const int nK = K / BK;
  issue(0, 0);
  for (int kk = 0; kk < nK; ++kk) {
    const int cur = kk & 1;
    if (kk + 1 < nK) {
      issue(cur ^ 1, kk + 1);                       // prefetch next tile
      asm volatile("s_wait_asynccnt 4" ::: "memory"); // this tile's 4 done
    } else {
      asm volatile("s_wait_asynccnt 0" ::: "memory");
    }
    __syncthreads();

    const int8_t* as = &As[cur][0];
    const int8_t* bs = &Bs[cur][0];
    v8i af[4], bf[2];
#pragma unroll
    for (int mt = 0; mt < 4; ++mt)
      af[mt] = frag_ld(as + (wm * 64 + mt * 16 + lrow) * AP, lhi);
#pragma unroll
    for (int nt = 0; nt < 2; ++nt)
      bf[nt] = frag_ld(bs + (wn * 32 + nt * 16 + lrow) * AP, lhi);
#pragma unroll
    for (int mt = 0; mt < 4; ++mt)
#pragma unroll
      for (int nt = 0; nt < 2; ++nt)
        acc[mt][nt] = __builtin_amdgcn_wmma_i32_16x16x64_iu8(
            ASIGN, af[mt], true, bf[nt], acc[mt][nt], false, false);
    __syncthreads();  // protect 'cur' buffer before next prefetch overwrites it
  }

  // Epilogue: C/D layout -> lane holds N = lane&15, VGPR r -> M = r + 8*(lane>>4).
  const float inv_s = 1.0f / s_next[0];
#pragma unroll
  for (int nt = 0; nt < 2; ++nt) {
    const int gn = bn + wn * 32 + nt * 16 + lrow;
    const float a = cA[gn];
    const float b = cB[gn];
#pragma unroll
    for (int mt = 0; mt < 4; ++mt) {
      const int gm0 = bm + wm * 64 + mt * 16 + lhi;
#pragma unroll
      for (int r = 0; r < 8; ++r) {
        float f = fmaf(a, (float)acc[mt][nt][r], b);
        f = fmaxf(f, 0.0f);
        int q = (int)rintf(f * inv_s);
        q = q < 0 ? 0 : (q > 15 ? 15 : q);
        Out[(size_t)(gm0 + r) * (size_t)ldo + (size_t)gn] = (int8_t)q;
      }
    }
  }
}

// Head GEMM: [4096,4096] x [16,4096]^T -> fp32 [4096,12]; one 16x16 WMMA tile
// per wave, K streamed straight from global (weights sit in L2).
__global__ __launch_bounds__(256) void gemm_iu8_out(
    const int8_t* __restrict__ A, const int8_t* __restrict__ Bq,
    const float* __restrict__ s3, const float* __restrict__ swo,
    float* __restrict__ Out, int K) {
  const int tid = threadIdx.x, lane = tid & 31, wave = tid >> 5;
  const int lrow = lane & 15, lhi = (lane >> 4) << 3;
  const int m0 = (blockIdx.x * 8 + wave) * 16;
  v8i acc = {0, 0, 0, 0, 0, 0, 0, 0};
  const int8_t* ap = A + (size_t)(m0 + lrow) * K;
  const int8_t* bp = Bq + (size_t)lrow * K;
  for (int k = 0; k < K; k += BK) {
    v8i a = frag_ld(ap + k, lhi);
    v8i b = frag_ld(bp + k, lhi);
    acc = __builtin_amdgcn_wmma_i32_16x16x64_iu8(false, a, true, b, acc,
                                                 false, false);
  }
  const float sc = s3[0] * swo[0];
  if (lrow < C_OUT) {
#pragma unroll
    for (int r = 0; r < 8; ++r)
      Out[(size_t)(m0 + lhi + r) * C_OUT + lrow] = sc * (float)acc[r];
  }
}

// ---------------------------------------------------------------------------
extern "C" void kernel_launch(void* const* d_in, const int* in_sizes, int n_in,
                              void* d_out, int out_size, void* d_ws, size_t ws_size,
                              hipStream_t stream) {
  (void)in_sizes; (void)n_in; (void)out_size; (void)ws_size;
  const float* x    = (const float*)d_in[0];
  const float* W1   = (const float*)d_in[1];
  const float* W2   = (const float*)d_in[2];
  const float* W3   = (const float*)d_in[3];
  const float* Wout = (const float*)d_in[4];
  const float* g1 = (const float*)d_in[5],  *b1 = (const float*)d_in[6];
  const float* m1 = (const float*)d_in[7],  *v1 = (const float*)d_in[8];
  const float* g2 = (const float*)d_in[9],  *b2 = (const float*)d_in[10];
  const float* m2 = (const float*)d_in[11], *v2 = (const float*)d_in[12];
  const float* g3 = (const float*)d_in[13], *b3 = (const float*)d_in[14];
  const float* m3 = (const float*)d_in[15], *v3 = (const float*)d_in[16];
  const float* s_in = (const float*)d_in[17], *s1 = (const float*)d_in[18];
  const float* s2   = (const float*)d_in[19], *s3 = (const float*)d_in[20];

  char* ws = (char*)d_ws;
  size_t off = 0;
  auto alloc = [&](size_t bytes) -> void* {
    off = (off + 255) & ~(size_t)255;
    void* p = ws + off;
    off += bytes;
    return p;
  };
  int8_t* w1q = (int8_t*)alloc((size_t)HID * F_PAD);
  int8_t* w2q = (int8_t*)alloc((size_t)HID * HID);
  int8_t* w3q = (int8_t*)alloc((size_t)HID * HID);
  int8_t* woq = (int8_t*)alloc((size_t)C_PAD * HID);
  int8_t* a0q = (int8_t*)alloc((size_t)B_SZ * F_PAD);
  int8_t* a1q = (int8_t*)alloc((size_t)B_SZ * HID);
  int8_t* a2q = (int8_t*)alloc((size_t)B_SZ * HID);
  float* sw1 = (float*)alloc(HID * 4);
  float* sw2 = (float*)alloc(HID * 4);
  float* sw3 = (float*)alloc(HID * 4);
  float* swo = (float*)alloc(4);
  float* cA1 = (float*)alloc(HID * 4); float* cB1 = (float*)alloc(HID * 4);
  float* cA2 = (float*)alloc(HID * 4); float* cB2 = (float*)alloc(HID * 4);
  float* cA3 = (float*)alloc(HID * 4); float* cB3 = (float*)alloc(HID * 4);

  // 1) quantize weights (4-bit per-channel; 8-bit per-tensor head) + input
  qw_rows<<<HID, 256, 0, stream>>>(W1, w1q, sw1, F_IN, F_PAD, 7.0f);
  qw_rows<<<HID, 256, 0, stream>>>(W2, w2q, sw2, HID, HID, 7.0f);
  qw_rows<<<HID, 256, 0, stream>>>(W3, w3q, sw3, HID, HID, 7.0f);
  qw_tensor<<<1, 256, 0, stream>>>(Wout, woq, swo, C_OUT, HID, C_PAD, 127.0f);
  qin<<<(B_SZ * F_PAD) / 256, 256, 0, stream>>>(x, a0q, s_in);

  // 2) fold BN + scales into per-channel epilogue coefficients
  coef<<<HID / 256, 256, 0, stream>>>(sw1, g1, b1, m1, v1, s_in, cA1, cB1);
  coef<<<HID / 256, 256, 0, stream>>>(sw2, g2, b2, m2, v2, s1, cA2, cB2);
  coef<<<HID / 256, 256, 0, stream>>>(sw3, g3, b3, m3, v3, s2, cA3, cB3);

  // 3) fused IU8-WMMA GEMM chain
  dim3 g(B_SZ / BM, HID / BN);
  gemm_iu8_bn_relu_q4<true ><<<g, 256, 0, stream>>>(a0q, w1q, cA1, cB1, s1, a1q, F_PAD, HID);
  gemm_iu8_bn_relu_q4<false><<<g, 256, 0, stream>>>(a1q, w2q, cA2, cB2, s2, a2q, HID, HID);
  gemm_iu8_bn_relu_q4<false><<<g, 256, 0, stream>>>(a2q, w3q, cA3, cB3, s3, a1q, HID, HID);
  gemm_iu8_out<<<B_SZ / 128, 256, 0, stream>>>(a1q, woq, s3, swo, (float*)d_out, HID);
}
